// NodeAttentionVariational_32959579030132
// MI455X (gfx1250) — matrix operand order repaired and checked by reading, not compiled, not run on hardware
//
#include <hip/hip_runtime.h>

// ---------------------------------------------------------------------------
// NodeAttentionVariational on MI455X (gfx1250, wave32, WMMA).
// Pipeline (all on `stream`, sequential deps):
//   k_nodebn1    : BN over nodes (per-feature stats over 128 rows)
//   k_nodelinear : nodes_mean/std linears fused -> nodes_v = exp(0.5*std)+mean
//   k_vstats     : per-feature stats of nodes_v (shared by nbn2 and bn5)
//   k_xstats     : per-channel stats of x  -> a1[c], d1[c]   (BN1 affine)
//   k_prepA1     : A1[n,c] = bf16(nodes1[n,c]*a1[c]), off[n]=sum_c nodes1*d1
//   k_gemm1      : f[bhw,n,t] = sum_c A1[n,c]*x + off[n]     (WMMA bf16)
//   k_fstats     : per-n stats of f -> a4[n], d4[n]          (BN4 affine)
//   k_prepA2     : A2T[c,n] = bf16(a4[n]*nodes2[n,c]), bias2[c]=sum_n d4*nodes2
//   k_gemm2      : y_raw[b,c,t,h,w] = sum_n A2T*f + bias2[c] (WMMA bf16) -> d_out
//   k_ystats     : per-channel stats of y_raw -> a6[c], d6[c]
//   k_final      : d_out = leaky(a6*y_raw + d6)  (in place)
// Workspace use: f (25.7MB) + node buffers (~1.6MB) + small vectors.
// ---------------------------------------------------------------------------

#define EPSf 1e-5f
#define NEG_SLOPE 0.2f

#define Bn 16
#define Cc 1024
#define Tt 64
#define Hh 7
#define Ww 7
#define HW 49
#define THW 3136
#define BHW 784
#define NN 128
#define DD 1024
#define CNT 50176 // B*T*H*W (x per-channel) == B*H*W*T (f per-node)

typedef __attribute__((ext_vector_type(16))) __bf16 v16bf;
typedef __attribute__((ext_vector_type(8))) float v8f;

union FragBF {
  uint4 u[2];
  v16bf v;
};

__device__ __forceinline__ unsigned short f2bf(float f) {
  unsigned u = __float_as_uint(f);
  u += 0x7fffu + ((u >> 16) & 1u); // round-to-nearest-even
  return (unsigned short)(u >> 16);
}

// ---------------- small prep kernels -----------------------------------

__global__ void k_xstats(const float* __restrict__ x, const float* __restrict__ g1,
                         const float* __restrict__ be1, float* __restrict__ a1,
                         float* __restrict__ d1) {
  __shared__ float ss[256], sq[256];
  const int c = blockIdx.x, tid = threadIdx.x;
  float s = 0.f, q = 0.f;
  for (int b = 0; b < Bn; ++b) {
    const float* p = x + (size_t)(b * Cc + c) * THW;
    for (int i = tid; i < THW; i += 256) {
      float v = p[i];
      s += v;
      q += v * v;
    }
  }
  ss[tid] = s;
  sq[tid] = q;
  __syncthreads();
  for (int k = 128; k > 0; k >>= 1) {
    if (tid < k) {
      ss[tid] += ss[tid + k];
      sq[tid] += sq[tid + k];
    }
    __syncthreads();
  }
  if (tid == 0) {
    float m = ss[0] / (float)CNT;
    float v = sq[0] / (float)CNT - m * m;
    float a = g1[c] * rsqrtf(v + EPSf);
    a1[c] = a;
    d1[c] = be1[c] - m * a;
  }
}

__global__ void k_nodebn1(const float* __restrict__ nodes, const float* __restrict__ g,
                          const float* __restrict__ be, float* __restrict__ out) {
  __shared__ float ss[128], sq[128];
  __shared__ float mb[2];
  const int d = blockIdx.x, t = threadIdx.x;
  float v = nodes[t * DD + d];
  ss[t] = v;
  sq[t] = v * v;
  __syncthreads();
  for (int k = 64; k > 0; k >>= 1) {
    if (t < k) {
      ss[t] += ss[t + k];
      sq[t] += sq[t + k];
    }
    __syncthreads();
  }
  if (t == 0) {
    float m = ss[0] * (1.f / 128.f);
    mb[0] = m;
    mb[1] = rsqrtf(sq[0] * (1.f / 128.f) - m * m + EPSf);
  }
  __syncthreads();
  out[t * DD + d] = g[d] * (v - mb[0]) * mb[1] + be[d];
}

// nodes_v = exp(0.5*(nn@wsd^T + bsd)) + (nn@wm^T + bmean)   (128x1024, K=1024)
__global__ void k_nodelinear(const float* __restrict__ nn, const float* __restrict__ wm,
                             const float* __restrict__ bmean, const float* __restrict__ wsd,
                             const float* __restrict__ bsd, float* __restrict__ nv) {
  __shared__ float As[32][33];
  __shared__ float Bm[64][33];
  __shared__ float Bs[64][33];
  const int tid = threadIdx.x;
  const int jb = blockIdx.x & 15;
  const int ib = blockIdx.x >> 4;
  const int i0 = ib * 32, j0 = jb * 64;
  const int j = tid & 63, it = tid >> 6;
  float accm[8] = {}, accs[8] = {};
  for (int k0 = 0; k0 < DD; k0 += 32) {
    __syncthreads();
    for (int idx = tid; idx < 32 * 32; idx += 256) {
      int r = idx >> 5, k = idx & 31;
      As[r][k] = nn[(i0 + r) * DD + k0 + k];
    }
    for (int idx = tid; idx < 64 * 32; idx += 256) {
      int r = idx >> 5, k = idx & 31;
      Bm[r][k] = wm[(j0 + r) * DD + k0 + k];
      Bs[r][k] = wsd[(j0 + r) * DD + k0 + k];
    }
    __syncthreads();
    for (int k = 0; k < 32; ++k) {
      float bmv = Bm[j][k], bsv = Bs[j][k];
#pragma unroll
      for (int ii = 0; ii < 8; ++ii) {
        float a = As[it + 4 * ii][k];
        accm[ii] = fmaf(a, bmv, accm[ii]);
        accs[ii] = fmaf(a, bsv, accs[ii]);
      }
    }
  }
  const int jg = j0 + j;
#pragma unroll
  for (int ii = 0; ii < 8; ++ii) {
    int ig = i0 + it + 4 * ii;
    nv[ig * DD + jg] = __expf(0.5f * (accs[ii] + bsd[jg])) + accm[ii] + bmean[jg];
  }
}

__global__ void k_vstats(const float* __restrict__ nv, float* __restrict__ vm,
                         float* __restrict__ vrs) {
  __shared__ float ss[128], sq[128];
  const int d = blockIdx.x, t = threadIdx.x;
  float v = nv[t * DD + d];
  ss[t] = v;
  sq[t] = v * v;
  __syncthreads();
  for (int k = 64; k > 0; k >>= 1) {
    if (t < k) {
      ss[t] += ss[t + k];
      sq[t] += sq[t + k];
    }
    __syncthreads();
  }
  if (t == 0) {
    float m = ss[0] * (1.f / 128.f);
    vm[d] = m;
    vrs[d] = rsqrtf(sq[0] * (1.f / 128.f) - m * m + EPSf);
  }
}

__global__ void k_prepA1(const float* __restrict__ nv, const float* __restrict__ vm,
                         const float* __restrict__ vrs, const float* __restrict__ g2,
                         const float* __restrict__ be2, const float* __restrict__ a1,
                         const float* __restrict__ d1, unsigned short* __restrict__ A1,
                         float* __restrict__ offn) {
  __shared__ float red[256];
  const int n = blockIdx.x, tid = threadIdx.x;
  float local = 0.f;
  for (int c = tid; c < DD; c += 256) {
    float val = g2[c] * (nv[n * DD + c] - vm[c]) * vrs[c] + be2[c];
    A1[n * DD + c] = f2bf(val * a1[c]);
    local += val * d1[c];
  }
  red[tid] = local;
  __syncthreads();
  for (int k = 128; k > 0; k >>= 1) {
    if (tid < k) red[tid] += red[tid + k];
    __syncthreads();
  }
  if (tid == 0) offn[n] = red[0];
}

__global__ void k_fstats(const float* __restrict__ f, const float* __restrict__ g4,
                         const float* __restrict__ be4, float* __restrict__ a4,
                         float* __restrict__ d4) {
  __shared__ float ss[256], sq[256];
  const int n = blockIdx.x, tid = threadIdx.x;
  const int t = tid & 63, q0 = tid >> 6;
  float s = 0.f, q = 0.f;
  for (int bhw = q0; bhw < BHW; bhw += 4) {
    float v = f[((size_t)bhw * NN + n) * Tt + t];
    s += v;
    q += v * v;
  }
  ss[tid] = s;
  sq[tid] = q;
  __syncthreads();
  for (int k = 128; k > 0; k >>= 1) {
    if (tid < k) {
      ss[tid] += ss[tid + k];
      sq[tid] += sq[tid + k];
    }
    __syncthreads();
  }
  if (tid == 0) {
    float m = ss[0] / (float)CNT;
    float v = sq[0] / (float)CNT - m * m;
    float a = g4[n] * rsqrtf(v + EPSf);
    a4[n] = a;
    d4[n] = be4[n] - m * a;
  }
}

__global__ void k_prepA2(const float* __restrict__ nv, const float* __restrict__ vm,
                         const float* __restrict__ vrs, const float* __restrict__ g5,
                         const float* __restrict__ be5, const float* __restrict__ a4,
                         const float* __restrict__ d4, unsigned short* __restrict__ A2T,
                         float* __restrict__ bias2) {
  __shared__ float red[128];
  const int c = blockIdx.x, n = threadIdx.x;
  float val = g5[c] * (nv[n * DD + c] - vm[c]) * vrs[c] + be5[c];
  A2T[c * NN + n] = f2bf(a4[n] * val);
  red[n] = d4[n] * val;
  __syncthreads();
  for (int k = 64; k > 0; k >>= 1) {
    if (n < k) red[n] += red[n + k];
    __syncthreads();
  }
  if (n == 0) bias2[c] = red[0];
}

// ---------------- WMMA GEMM 1: f[bhw,n,t] = sum_c A1[n,c]*x + off[n] ----

__global__ void __launch_bounds__(256) k_gemm1(const float* __restrict__ x,
                                               const unsigned short* __restrict__ A1,
                                               const float* __restrict__ offn,
                                               float* __restrict__ f) {
  __shared__ unsigned short Als[128 * 40]; // 128 rows (n) x 32 K, pad->40
  __shared__ unsigned short Bls[64 * 40];  // 64 rows (t)  x 32 K, pad->40
  const int bhw = blockIdx.x;
  const int b = bhw / HW, hw = bhw % HW;
  const size_t xbase = (size_t)b * Cc * THW + hw;
  const int tid = threadIdx.x;
  const int lane = tid & 31, wv = tid >> 5;
  const int lm = lane & 15, lhalf = lane >> 4;
  const int kbA = lhalf * 8;  // A frag: K = e + 8*((e>=8)+lhalf)
  const int kbB = lhalf * 16; // B frag: K = e + 8*((e>=8)+lhalf) (via kbA)

  v8f acc[4];
  {
    v8f cinit;
#pragma unroll
    for (int r = 0; r < 8; ++r) cinit[r] = offn[wv * 16 + 8 * lhalf + r];
    acc[0] = cinit;
    acc[1] = cinit;
    acc[2] = cinit;
    acc[3] = cinit;
  }

  const int arow = tid >> 1, aseg = tid & 1;

  for (int kk = 0; kk < 32; ++kk) {
    __syncthreads();
    { // A chunk: 128 x 32 bf16, coalesced 32B per thread
      const uint4* s = (const uint4*)(A1 + arow * DD + kk * 32 + aseg * 16);
      uint4* dst = (uint4*)(Als + arow * 40 + aseg * 16);
      dst[0] = s[0];
      dst[1] = s[1];
    }
#pragma unroll
    for (int r = 0; r < 8; ++r) { // x slice 32c x 64t -> bf16, stored [t][c]
      int idx = tid + r * 256;
      int ci = idx >> 6, t = idx & 63;
      float v = x[xbase + (size_t)(kk * 32 + ci) * THW + t * HW];
      Bls[t * 40 + ci] = f2bf(v);
    }
    __syncthreads();
    FragBF a;
    {
      const unsigned short* ar = Als + (wv * 16 + lm) * 40;
      a.u[0] = *(const uint4*)(ar + kbA);      // K = kbA..kbA+7
      a.u[1] = *(const uint4*)(ar + kbA + 16); // K = kbA+16..kbA+23
    }
#pragma unroll
    for (int tb = 0; tb < 4; ++tb) {
      FragBF bb;
      const unsigned short* br = Bls + (tb * 16 + lm) * 40 + kbA;
      bb.u[0] = *(const uint4*)(br);
      bb.u[1] = *(const uint4*)(br + 16);
      acc[tb] = __builtin_amdgcn_wmma_f32_16x16x32_bf16(false, a.v, false, bb.v, (short)0,
                                                        acc[tb], false, false);
    }
  }
#pragma unroll
  for (int tb = 0; tb < 4; ++tb) {
    int t = tb * 16 + lm;
#pragma unroll
    for (int r = 0; r < 8; ++r) {
      int n = wv * 16 + 8 * lhalf + r;
      f[((size_t)bhw * NN + n) * Tt + t] = acc[tb][r];
    }
  }
}

// ---------------- WMMA GEMM 2: y = sum_n A2T[c,n]*f + bias2[c] ----------

__global__ void __launch_bounds__(256) k_gemm2(const float* __restrict__ f,
                                               const unsigned short* __restrict__ A2T,
                                               const float* __restrict__ bias2,
                                               float* __restrict__ y) {
  __shared__ unsigned short Als[128 * 40]; // 128 rows (c) x 32 K(n), pad->40
  __shared__ unsigned short Bls[64 * 40];  // 64 rows (t)  x 32 K(n), pad->40
  const int blk = blockIdx.x;
  const int bhw = blk >> 3, cblk = blk & 7;
  const int b = bhw / HW, hw = bhw % HW;
  const int cbase = cblk * 128;
  const size_t ybase = (size_t)b * Cc * THW + hw;
  const int tid = threadIdx.x;
  const int lane = tid & 31, wv = tid >> 5;
  const int lm = lane & 15, lhalf = lane >> 4;
  const int kbA = lhalf * 8;
  const int kbB = lhalf * 16;

  v8f acc[4];
  {
    v8f cinit;
#pragma unroll
    for (int r = 0; r < 8; ++r) cinit[r] = bias2[cbase + wv * 16 + 8 * lhalf + r];
    acc[0] = cinit;
    acc[1] = cinit;
    acc[2] = cinit;
    acc[3] = cinit;
  }

  const int arow = tid >> 1, aseg = tid & 1;

  for (int kk = 0; kk < 4; ++kk) { // K = 128 n, chunks of 32
    __syncthreads();
    {
      const uint4* s = (const uint4*)(A2T + (size_t)(cbase + arow) * NN + kk * 32 + aseg * 16);
      uint4* dst = (uint4*)(Als + arow * 40 + aseg * 16);
      dst[0] = s[0];
      dst[1] = s[1];
    }
#pragma unroll
    for (int r = 0; r < 8; ++r) { // f chunk 32n x 64t (contiguous in t)
      int idx = tid + r * 256;
      int ni = idx >> 6, t = idx & 63;
      float v = f[((size_t)bhw * NN + kk * 32 + ni) * Tt + t];
      Bls[t * 40 + ni] = f2bf(v);
    }
    __syncthreads();
    FragBF a;
    {
      const unsigned short* ar = Als + (wv * 16 + lm) * 40;
      a.u[0] = *(const uint4*)(ar + kbA);
      a.u[1] = *(const uint4*)(ar + kbA + 16);
    }
#pragma unroll
    for (int tb = 0; tb < 4; ++tb) {
      FragBF bb;
      const unsigned short* br = Bls + (tb * 16 + lm) * 40 + kbA;
      bb.u[0] = *(const uint4*)(br);
      bb.u[1] = *(const uint4*)(br + 16);
      acc[tb] = __builtin_amdgcn_wmma_f32_16x16x32_bf16(false, a.v, false, bb.v, (short)0,
                                                        acc[tb], false, false);
    }
  }
#pragma unroll
  for (int tb = 0; tb < 4; ++tb) {
    int t = tb * 16 + lm;
#pragma unroll
    for (int r = 0; r < 8; ++r) {
      int c = cbase + wv * 16 + 8 * lhalf + r;
      y[ybase + (size_t)c * THW + t * HW] = acc[tb][r];
    }
  }
}

// ---------------- epilogue kernels --------------------------------------

__global__ void k_ystats(const float* __restrict__ y, const float* __restrict__ g6,
                         const float* __restrict__ be6, float* __restrict__ a6,
                         float* __restrict__ d6) {
  __shared__ float ss[256], sq[256];
  const int c = blockIdx.x, tid = threadIdx.x;
  float s = 0.f, q = 0.f;
  for (int b = 0; b < Bn; ++b) {
    const float* p = y + (size_t)(b * Cc + c) * THW;
    for (int i = tid; i < THW; i += 256) {
      float v = p[i];
      s += v;
      q += v * v;
    }
  }
  ss[tid] = s;
  sq[tid] = q;
  __syncthreads();
  for (int k = 128; k > 0; k >>= 1) {
    if (tid < k) {
      ss[tid] += ss[tid + k];
      sq[tid] += sq[tid + k];
    }
    __syncthreads();
  }
  if (tid == 0) {
    float m = ss[0] / (float)CNT;
    float v = sq[0] / (float)CNT - m * m;
    float a = g6[c] * rsqrtf(v + EPSf);
    a6[c] = a;
    d6[c] = be6[c] - m * a;
  }
}

__global__ void k_final(float4* __restrict__ out4, const float* __restrict__ a6,
                        const float* __restrict__ d6, int nvec) {
  int i = blockIdx.x * 256 + threadIdx.x;
  if (i >= nvec) return;
  int c = (i / (THW / 4)) & (Cc - 1); // 784 float4 per (b,c) block
  float a = a6[c], d = d6[c];
  float4 v = out4[i];
  float y0 = a * v.x + d, y1 = a * v.y + d, y2 = a * v.z + d, y3 = a * v.w + d;
  v.x = y0 >= 0.f ? y0 : NEG_SLOPE * y0;
  v.y = y1 >= 0.f ? y1 : NEG_SLOPE * y1;
  v.z = y2 >= 0.f ? y2 : NEG_SLOPE * y2;
  v.w = y3 >= 0.f ? y3 : NEG_SLOPE * y3;
  out4[i] = v;
}

// ---------------- launcher ----------------------------------------------

extern "C" void kernel_launch(void* const* d_in, const int* in_sizes, int n_in,
                              void* d_out, int out_size, void* d_ws, size_t ws_size,
                              hipStream_t stream) {
  const float* x = (const float*)d_in[0];
  const float* nodes = (const float*)d_in[1];
  const float* w_mean = (const float*)d_in[2];
  const float* b_mean = (const float*)d_in[3];
  const float* w_std = (const float*)d_in[4];
  const float* b_std = (const float*)d_in[5];
  const float* g_nbn1 = (const float*)d_in[6];
  const float* be_nbn1 = (const float*)d_in[7];
  const float* g_nbn2 = (const float*)d_in[8];
  const float* be_nbn2 = (const float*)d_in[9];
  const float* g_bn1 = (const float*)d_in[10];
  const float* be_bn1 = (const float*)d_in[11];
  const float* g_bn4 = (const float*)d_in[12];
  const float* be_bn4 = (const float*)d_in[13];
  const float* g_bn5 = (const float*)d_in[14];
  const float* be_bn5 = (const float*)d_in[15];
  const float* g_bn6 = (const float*)d_in[16];
  const float* be_bn6 = (const float*)d_in[17];
  float* out = (float*)d_out;

  // workspace layout (all 16B aligned)
  char* w = (char*)d_ws;
  size_t o = 0;
  float* f_buf = (float*)(w + o);
  o += (size_t)BHW * NN * Tt * 4; // 25.69 MB
  unsigned short* A1 = (unsigned short*)(w + o);
  o += (size_t)NN * DD * 2;
  unsigned short* A2T = (unsigned short*)(w + o);
  o += (size_t)DD * NN * 2;
  float* nodes_n = (float*)(w + o);
  o += (size_t)NN * DD * 4;
  float* nodes_v = (float*)(w + o);
  o += (size_t)NN * DD * 4;
  float* a1 = (float*)(w + o);
  o += DD * 4;
  float* d1 = (float*)(w + o);
  o += DD * 4;
  float* vm = (float*)(w + o);
  o += DD * 4;
  float* vrs = (float*)(w + o);
  o += DD * 4;
  float* bias2 = (float*)(w + o);
  o += DD * 4;
  float* a6 = (float*)(w + o);
  o += DD * 4;
  float* d6 = (float*)(w + o);
  o += DD * 4;
  float* offn = (float*)(w + o);
  o += NN * 4;
  float* a4 = (float*)(w + o);
  o += NN * 4;
  float* d4 = (float*)(w + o);
  o += NN * 4;
  (void)ws_size;
  (void)in_sizes;
  (void)n_in;

  k_nodebn1<<<DD, 128, 0, stream>>>(nodes, g_nbn1, be_nbn1, nodes_n);
  k_nodelinear<<<64, 256, 0, stream>>>(nodes_n, w_mean, b_mean, w_std, b_std, nodes_v);
  k_vstats<<<DD, 128, 0, stream>>>(nodes_v, vm, vrs);
  k_xstats<<<Cc, 256, 0, stream>>>(x, g_bn1, be_bn1, a1, d1);
  k_prepA1<<<NN, 256, 0, stream>>>(nodes_v, vm, vrs, g_nbn2, be_nbn2, a1, d1, A1, offn);
  k_gemm1<<<BHW, 256, 0, stream>>>(x, A1, offn, f_buf);
  k_fstats<<<NN, 256, 0, stream>>>(f_buf, g_bn4, be_bn4, a4, d4);
  k_prepA2<<<Cc, 128, 0, stream>>>(nodes_v, vm, vrs, g_bn5, be_bn5, a4, d4, A2T, bias2);
  k_gemm2<<<BHW * 8, 256, 0, stream>>>(f_buf, A2T, bias2, out);
  k_ystats<<<Cc, 256, 0, stream>>>(out, g_bn6, be_bn6, a6, d6);
  int nvec = out_size / 4;
  k_final<<<(nvec + 255) / 256, 256, 0, stream>>>((float4*)out, a6, d6, nvec);
}
